// DeepseekV3MLP_52785148067899
// MI455X (gfx1250) — compile-verified, matrix operand order checked by reading, not run on hardware
//
#include <hip/hip_runtime.h>

// ---------------------------------------------------------------------------
// SwiGLU MLP with block-dequantized fp32 weights, bf16 WMMA on gfx1250.
//   T=512, H=4096, I=11008, BS=128
// Kernel 1: h = silu(x@Wg^T) * (x@Wu^T)   -> bf16 workspace [T, I]
// Kernel 2: out = h @ Wd                  -> fp32 [T, H]
// HBM-bound (~541MB weights). Double-buffered LDS tiles, software-pipelined:
// issue global loads -> WMMA on current tile -> convert/store next tile.
// Kernel 2's bf16 A-tile uses global_load_async_to_lds_b128 (ASYNCcnt path).
// ---------------------------------------------------------------------------

typedef __attribute__((ext_vector_type(16))) __bf16          v16bf;
typedef __attribute__((ext_vector_type(8)))  float           v8f;
typedef __attribute__((ext_vector_type(4)))  unsigned int    u32x4;
typedef __attribute__((ext_vector_type(4)))  unsigned short  u16x4;

constexpr int T = 512;
constexpr int H = 4096;
constexpr int I = 11008;
constexpr int ROWB  = 80;          // LDS row stride: 32 bf16 (64B) + 16B pad
constexpr int TILEB = 128 * ROWB;  // 10240 B per tile

union FragBF { v16bf v; u32x4 u[2]; };

__device__ __forceinline__ unsigned short f2bf(float f) {
  union { __bf16 b; unsigned short s; } c;
  c.b = (__bf16)f;                 // native cvt (v_cvt_pk_bf16_f32 class)
  return c.s;
}

// Fragment load matching the ISA bf16 16x32 A / 32x16 B layout:
// lane&15 = M (or N) row; kh = lane>>4 selects the K-half.
// VGPRs 0..3 <- K = kh*8..kh*8+7 ; VGPRs 4..7 <- K = 16+kh*8..16+kh*8+7.
__device__ __forceinline__ FragBF load_frag(const unsigned char* lds, int row, int kh) {
  FragBF f;
  const unsigned char* p = lds + row * ROWB;
  f.u[0] = *(const u32x4*)(p + kh * 16);
  f.u[1] = *(const u32x4*)(p + 32 + kh * 16);
  return f;
}

// Low 32 bits of a generic pointer to LDS == byte address within LDS aperture.
__device__ __forceinline__ unsigned lds_off32(const void* p) {
  return (unsigned)(uintptr_t)p;
}

// CDNA5 async copy: 16B per lane, global -> LDS, tracked by ASYNCcnt.
__device__ __forceinline__ void async_load_b128(const void* lds_dst,
                                                const void* gbase,
                                                unsigned byte_off) {
  asm volatile("global_load_async_to_lds_b128 %0, %1, %2"
               :: "v"(lds_off32(lds_dst)), "v"(byte_off), "s"(gbase)
               : "memory");
}
__device__ __forceinline__ void wait_async0() {
  asm volatile("s_wait_asynccnt 0x0" ::: "memory");
}

// ---------------------------------------------------------------------------
// Kernel 1: fused gate/up GEMM + SiLU*up, 128(T) x 128(I) tile per block.
// ---------------------------------------------------------------------------
__global__ __launch_bounds__(256, 1)
void k_gateup(const float* __restrict__ x,   const float* __restrict__ wgp,
              const float* __restrict__ wup, const float* __restrict__ gsc,
              const float* __restrict__ usc, unsigned short* __restrict__ hact) {
  __shared__ __align__(16) unsigned char ldsX[2][TILEB];
  __shared__ __align__(16) unsigned char ldsG[2][TILEB];
  __shared__ __align__(16) unsigned char ldsU[2][TILEB];

  const int i0   = blockIdx.x * 128;
  const int t0   = blockIdx.y * 128;
  const int tid  = threadIdx.x;
  const int lane = tid & 31;
  const int wave = tid >> 5;
  const int mrow = wave * 16 + (lane & 15);
  const int nrow = lane & 15;
  const int kh   = lane >> 4;
  const int srow = (i0 >> 7) * (H >> 7);

  float4 rx[4], rg[4], ru[4];   // staged next-tile loads (register pipeline)

  auto issue = [&](int k0) {
#pragma unroll
    for (int r = 0; r < 4; ++r) {
      const int idx = tid + r * 256, row = idx >> 3, c4 = idx & 7;
      rx[r] = *(const float4*)(x   + (size_t)(t0 + row) * H + k0 + c4 * 4);
      rg[r] = *(const float4*)(wgp + (size_t)(i0 + row) * H + k0 + c4 * 4);
      ru[r] = *(const float4*)(wup + (size_t)(i0 + row) * H + k0 + c4 * 4);
    }
  };
  auto stage = [&](int buf, int k0) {
    const float sg = gsc[srow + (k0 >> 7)];
    const float su = usc[srow + (k0 >> 7)];
#pragma unroll
    for (int r = 0; r < 4; ++r) {
      const int idx = tid + r * 256, row = idx >> 3, c4 = idx & 7;
      u16x4 bx = { f2bf(rx[r].x),      f2bf(rx[r].y),      f2bf(rx[r].z),      f2bf(rx[r].w)      };
      u16x4 bg = { f2bf(rg[r].x * sg), f2bf(rg[r].y * sg), f2bf(rg[r].z * sg), f2bf(rg[r].w * sg) };
      u16x4 bu = { f2bf(ru[r].x * su), f2bf(ru[r].y * su), f2bf(ru[r].z * su), f2bf(ru[r].w * su) };
      *(u16x4*)(ldsX[buf] + row * ROWB + c4 * 8) = bx;
      *(u16x4*)(ldsG[buf] + row * ROWB + c4 * 8) = bg;
      *(u16x4*)(ldsU[buf] + row * ROWB + c4 * 8) = bu;
    }
  };

  v8f accG[8], accU[8];
#pragma unroll
  for (int n = 0; n < 8; ++n) { accG[n] = {}; accU[n] = {}; }

  issue(0);
  stage(0, 0);
  __syncthreads();

  for (int k0 = 0, it = 0; k0 < H; k0 += 32, ++it) {
    const int  cur  = it & 1;
    const bool more = (k0 + 32) < H;
    if (more) issue(k0 + 32);                 // loads in flight during WMMA phase

    const FragBF a = load_frag(ldsX[cur], mrow, kh);
#pragma unroll
    for (int nt = 0; nt < 8; ++nt) {
      const FragBF bg = load_frag(ldsG[cur], nt * 16 + nrow, kh);
      accG[nt] = __builtin_amdgcn_wmma_f32_16x16x32_bf16(
          false, a.v, false, bg.v, (short)0, accG[nt], false, false);
      const FragBF bu = load_frag(ldsU[cur], nt * 16 + nrow, kh);
      accU[nt] = __builtin_amdgcn_wmma_f32_16x16x32_bf16(
          false, a.v, false, bu.v, (short)0, accU[nt], false, false);
    }
    if (more) stage(cur ^ 1, k0 + 32);        // convert + LDS store after compute
    __syncthreads();
  }

  // Epilogue: silu(gate)*up -> bf16. C/D layout: lane&15 = N, (lane>>4)*8+j = M.
  const int tbase = t0 + wave * 16 + (lane >> 4) * 8;
  const int ibase = i0 + (lane & 15);
#pragma unroll
  for (int nt = 0; nt < 8; ++nt) {
#pragma unroll
    for (int j = 0; j < 8; ++j) {
      const float g = accG[nt][j];
      const float u = accU[nt][j];
      const float hval = (g / (1.0f + __expf(-g))) * u;
      hact[(size_t)(tbase + j) * I + ibase + nt * 16] = f2bf(hval);
    }
  }
}

// ---------------------------------------------------------------------------
// Kernel 2: out = h @ Wd, 128(T) x 128(H) tile per block.
// A-tile (bf16, no conversion) copied with global_load_async_to_lds_b128.
// ---------------------------------------------------------------------------
__global__ __launch_bounds__(256, 1)
void k_down(const unsigned short* __restrict__ hact, const float* __restrict__ wdp,
            const float* __restrict__ dsc, float* __restrict__ out) {
  __shared__ __align__(16) unsigned char ldsA[2][TILEB];
  __shared__ __align__(16) unsigned char ldsB[2][TILEB];

  const int h0   = blockIdx.x * 128;
  const int t0   = blockIdx.y * 128;
  const int tid  = threadIdx.x;
  const int lane = tid & 31;
  const int wave = tid >> 5;
  const int mrow = wave * 16 + (lane & 15);
  const int nrow = lane & 15;
  const int kh   = lane >> 4;

  float4 rb[4];   // staged Wd tile

  auto issueA = [&](int buf, int k0) {        // async DMA: bf16 h tile -> LDS
#pragma unroll
    for (int r = 0; r < 2; ++r) {
      const int idx = tid + r * 256, row = idx >> 2, c16 = idx & 3;
      const unsigned goff = (unsigned)(((size_t)(t0 + row) * I + k0 + c16 * 8) * 2u);
      async_load_b128(ldsA[buf] + row * ROWB + c16 * 16, hact, goff);
    }
  };
  auto issueB = [&](int k0) {
#pragma unroll
    for (int r = 0; r < 4; ++r) {
      const int idx = tid + r * 256, krow = idx >> 5, c4 = idx & 31;
      rb[r] = *(const float4*)(wdp + (size_t)(k0 + krow) * H + h0 + c4 * 4);
    }
  };
  auto stageB = [&](int buf, int k0) {        // dequant + transpose into ldsB[n][k]
    const float sd = dsc[(k0 >> 7) * (H >> 7) + (h0 >> 7)];
#pragma unroll
    for (int r = 0; r < 4; ++r) {
      const int idx = tid + r * 256, krow = idx >> 5, c4 = idx & 31;
      const int n = c4 * 4;
      *(unsigned short*)(ldsB[buf] + (n + 0) * ROWB + krow * 2) = f2bf(rb[r].x * sd);
      *(unsigned short*)(ldsB[buf] + (n + 1) * ROWB + krow * 2) = f2bf(rb[r].y * sd);
      *(unsigned short*)(ldsB[buf] + (n + 2) * ROWB + krow * 2) = f2bf(rb[r].z * sd);
      *(unsigned short*)(ldsB[buf] + (n + 3) * ROWB + krow * 2) = f2bf(rb[r].w * sd);
    }
  };

  v8f acc[8];
#pragma unroll
  for (int n = 0; n < 8; ++n) acc[n] = {};

  issueA(0, 0);
  issueB(0);
  stageB(0, 0);
  wait_async0();
  __syncthreads();

  for (int k0 = 0, it = 0; k0 < I; k0 += 32, ++it) {
    const int  cur  = it & 1;
    const bool more = (k0 + 32) < I;
    if (more) { issueA(cur ^ 1, k0 + 32); issueB(k0 + 32); }

    const FragBF a = load_frag(ldsA[cur], mrow, kh);
#pragma unroll
    for (int nt = 0; nt < 8; ++nt) {
      const FragBF b = load_frag(ldsB[cur], nt * 16 + nrow, kh);
      acc[nt] = __builtin_amdgcn_wmma_f32_16x16x32_bf16(
          false, a.v, false, b.v, (short)0, acc[nt], false, false);
    }
    if (more) stageB(cur ^ 1, k0 + 32);
    wait_async0();                            // this wave's async LDS writes done
    __syncthreads();                          // -> visible to all waves
  }

  const int tbase = t0 + wave * 16 + (lane >> 4) * 8;
  const int hbase = h0 + (lane & 15);
#pragma unroll
  for (int nt = 0; nt < 8; ++nt) {
#pragma unroll
    for (int j = 0; j < 8; ++j) {
      out[(size_t)(tbase + j) * H + hbase + nt * 16] = acc[nt][j];
    }
  }
}

// ---------------------------------------------------------------------------
extern "C" void kernel_launch(void* const* d_in, const int* in_sizes, int n_in,
                              void* d_out, int out_size, void* d_ws, size_t ws_size,
                              hipStream_t stream) {
  const float* x   = (const float*)d_in[0];
  const float* wg  = (const float*)d_in[1];
  const float* wu  = (const float*)d_in[2];
  const float* wd  = (const float*)d_in[3];
  const float* gsc = (const float*)d_in[4];
  const float* usc = (const float*)d_in[5];
  const float* dsc = (const float*)d_in[6];
  // d_in[7] = blocksize (int, ==128) — baked into the tiling.

  float* out = (float*)d_out;
  unsigned short* hact = (unsigned short*)d_ws;   // bf16 [T, I] intermediate

  dim3 blk(256);
  dim3 grd1(I / 128, T / 128);   // 86 x 4
  dim3 grd2(H / 128, T / 128);   // 32 x 4
  k_gateup<<<grd1, blk, 0, stream>>>(x, wg, wu, gsc, usc, hact);
  k_down  <<<grd2, blk, 0, stream>>>(hact, wd, dsc, out);
}
